// dozerformer_Decoder_26508538151378
// MI455X (gfx1250) — matrix operand order repaired and checked by reading, not compiled, not run on hardware
//
#include <hip/hip_runtime.h>
#include <math.h>

// ---------------- static problem config (matches reference) ----------------
namespace cfg {
constexpr int Bsz = 32, Cc = 21, Pp = 24, Ee = 64;
constexpr int Sdec = 28, Senc = 30, Ldec = 672;
constexpr int Dm = 1536, Fm = 3072, Hh = 8, Dh = 192;   // d_model, d_ff, heads, head_dim
constexpr int BC = Bsz * Cc;          // 672 sequences
constexpr int Mx = BC * Sdec;         // 18816 decoder-side rows
constexpr int Mc = BC * Senc;         // 20160 encoder-side rows
constexpr int LOCAL = 3, STRIDE = 4;
}

// ---------------- types for WMMA bf16 ----------------
typedef __bf16 v16bf __attribute__((ext_vector_type(16)));
typedef float  v8f   __attribute__((ext_vector_type(8)));
typedef unsigned int u32x4 __attribute__((ext_vector_type(4)));

union Frag16 { v16bf m; u32x4 u[2]; };

// ---------------- helpers ----------------
static __device__ __forceinline__ unsigned short f2b(float f) {
    // f32 -> bf16 round-to-nearest-even
    unsigned int u = __float_as_uint(f);
    unsigned int r = u + 0x7FFFu + ((u >> 16) & 1u);
    return (unsigned short)(r >> 16);
}
static __device__ __forceinline__ float b2f(unsigned short h) {
    return __uint_as_float(((unsigned int)h) << 16);
}
static __device__ __forceinline__ float gelu_tanh(float x) {
    float x3 = x * x * x;
    return 0.5f * x * (1.0f + tanhf(0.7978845608028654f * (x + 0.044715f * x3)));
}
// Async DMA global -> LDS, 16 bytes, no VGPR data path (ASYNCcnt tracked).
static __device__ __forceinline__ void async_to_lds_b128(unsigned lds_byte_off,
                                                         const void* gaddr) {
    asm volatile("global_load_async_to_lds_b128 %0, %1, off"
                 :: "v"(lds_byte_off), "v"(gaddr) : "memory");
}
static __device__ __forceinline__ void wait_asynccnt0() {
    asm volatile("s_wait_asynccnt 0x0" ::: "memory");
}

// ============================================================================
// Kernel 1: weight transpose + f32->bf16 convert.  Wt[n*K+k] = bf16(W[k*N+n])
// ============================================================================
__global__ __launch_bounds__(256)
void convert_wt(const float* __restrict__ W, unsigned short* __restrict__ Wt,
                int K, int N) {
    size_t t = (size_t)blockIdx.x * blockDim.x + threadIdx.x;
    if (t >= (size_t)N * K) return;
    int n = (int)(t / K), k = (int)(t % K);
    Wt[t] = f2b(W[(size_t)k * N + n]);
}

// ============================================================================
// Kernel 2: build X = rearranged(value_emb(x_dec) + pos)  -> (BC, Sdec, D) f32
// ============================================================================
__global__ __launch_bounds__(256)
void build_x0(const float* __restrict__ x_dec, const float* __restrict__ val_w,
              const float* __restrict__ val_b, const float* __restrict__ pos,
              float* __restrict__ X) {
    using namespace cfg;
    size_t t = (size_t)blockIdx.x * blockDim.x + threadIdx.x;
    if (t >= (size_t)Mx * Dm) return;
    int d = (int)(t % Dm);
    int row = (int)(t / Dm);
    int s = row % Sdec, bc = row / Sdec;
    int b = bc / Cc, c = bc % Cc;
    int p = d / Ee, e = d % Ee;
    int l = s * Pp + p;
    float emb = x_dec[((size_t)b * Ldec + l) * Cc + c] * val_w[e] + val_b[e];
    float ps  = pos[(((size_t)e * Sdec + s) * Pp + p) * Cc + c];
    X[t] = emb + ps;
}

// ============================================================================
// Kernel 3: rearrange cross -> bf16 (BC, Senc, D)
// ============================================================================
__global__ __launch_bounds__(256)
void build_xc(const float* __restrict__ cross, unsigned short* __restrict__ XCh) {
    using namespace cfg;
    size_t t = (size_t)blockIdx.x * blockDim.x + threadIdx.x;
    if (t >= (size_t)Mc * Dm) return;
    int d = (int)(t % Dm);
    int row = (int)(t / Dm);
    int s = row % Senc, bc = row / Senc;
    int b = bc / Cc, c = bc % Cc;
    int p = d / Ee, e = d % Ee;
    XCh[t] = f2b(cross[((((size_t)b * Ee + e) * Senc + s) * Pp + p) * Cc + c]);
}

// ============================================================================
// Kernel 4: row layernorm over D=1536, in-place f32, optional bf16 shadow
// ============================================================================
__global__ __launch_bounds__(256)
void layernorm_rows(float* __restrict__ X, const float* __restrict__ g,
                    const float* __restrict__ bgain, unsigned short* __restrict__ outH) {
    using namespace cfg;
    __shared__ float red[2][8];
    const int tid = threadIdx.x, lane = tid & 31, wid = tid >> 5;
    float* row = X + (size_t)blockIdx.x * Dm;
    float s = 0.f, sq = 0.f;
    for (int d = tid; d < Dm; d += 256) { float x = row[d]; s += x; sq += x * x; }
    #pragma unroll
    for (int o = 16; o; o >>= 1) { s += __shfl_xor(s, o, 32); sq += __shfl_xor(sq, o, 32); }
    if (lane == 0) { red[0][wid] = s; red[1][wid] = sq; }
    __syncthreads();
    if (wid == 0) {
        s  = (lane < 8) ? red[0][lane] : 0.f;
        sq = (lane < 8) ? red[1][lane] : 0.f;
        #pragma unroll
        for (int o = 4; o; o >>= 1) { s += __shfl_xor(s, o, 32); sq += __shfl_xor(sq, o, 32); }
        if (lane == 0) { red[0][0] = s; red[1][0] = sq; }
    }
    __syncthreads();
    s = red[0][0]; sq = red[1][0];
    const float mean = s * (1.0f / Dm);
    const float var  = sq * (1.0f / Dm) - mean * mean;
    const float rstd = rsqrtf(var + 1e-5f);
    unsigned short* hrow = outH ? outH + (size_t)blockIdx.x * Dm : nullptr;
    for (int d = tid; d < Dm; d += 256) {
        float y = (row[d] - mean) * rstd * g[d] + bgain[d];
        row[d] = y;
        if (hrow) hrow[d] = f2b(y);
    }
}

// ============================================================================
// Kernel 5: WMMA bf16 GEMM  C[M,N] = A[M,K] @ Bt[N,K]^T + bias
//   mode 0: outF[m,n] += result            (fused residual, in place)
//   mode 1: outH[m,n]  = bf16(result)
//   mode 2: outH[m,n]  = bf16(gelu(result))
// Each wave computes a 16(M) x 64(N) strip; K stepped by 32.
// Fragment k order is contiguous: lane half hi holds k = 16*hi + s in A and B alike.
// ============================================================================
__global__ __launch_bounds__(256)
void gemm_bf16_wmma(const unsigned short* __restrict__ A,
                    const unsigned short* __restrict__ Bt,
                    const float* __restrict__ bias,
                    float* __restrict__ outF,
                    unsigned short* __restrict__ outH,
                    int M, int N, int K, int mode) {
    const int waveId = blockIdx.x * (blockDim.x >> 5) + (threadIdx.x >> 5);
    const int lane   = threadIdx.x & 31;
    const int tilesN = N >> 6;               // 64 columns per wave
    const int nTiles = (M >> 4) * tilesN;
    if (waveId >= nTiles) return;            // whole-wave uniform exit (EXEC stays full)
    const int tM = waveId / tilesN;
    const int tN = waveId % tilesN;
    const int l15 = lane & 15;
    const int hi  = lane >> 4;

    const unsigned short* aRow  = A  + (size_t)(tM * 16 + l15) * K + hi * 16;
    const unsigned short* bBase = Bt + (size_t)(tN * 64 + l15) * K + hi * 16;
    const size_t bStride = (size_t)16 * K;   // +16 output columns

    v8f acc0 = {}, acc1 = {}, acc2 = {}, acc3 = {};
    for (int kk = 0; kk < K; kk += 32) {
        __builtin_prefetch(aRow + kk + 128, 0, 1);
        __builtin_prefetch(bBase + kk + 128, 0, 1);
        Frag16 a, b0, b1, b2, b3;
        a.u[0] = *(const u32x4*)(aRow + kk);
        a.u[1] = *(const u32x4*)(aRow + kk + 8);
        const unsigned short* bp = bBase + kk;
        b0.u[0] = *(const u32x4*)(bp);                  b0.u[1] = *(const u32x4*)(bp + 8);
        b1.u[0] = *(const u32x4*)(bp + bStride);        b1.u[1] = *(const u32x4*)(bp + bStride + 8);
        b2.u[0] = *(const u32x4*)(bp + 2 * bStride);    b2.u[1] = *(const u32x4*)(bp + 2 * bStride + 8);
        b3.u[0] = *(const u32x4*)(bp + 3 * bStride);    b3.u[1] = *(const u32x4*)(bp + 3 * bStride + 8);
        acc0 = __builtin_amdgcn_wmma_f32_16x16x32_bf16(false, a.m, false, b0.m, (short)0, acc0, false, false);
        acc1 = __builtin_amdgcn_wmma_f32_16x16x32_bf16(false, a.m, false, b1.m, (short)0, acc1, false, false);
        acc2 = __builtin_amdgcn_wmma_f32_16x16x32_bf16(false, a.m, false, b2.m, (short)0, acc2, false, false);
        acc3 = __builtin_amdgcn_wmma_f32_16x16x32_bf16(false, a.m, false, b3.m, (short)0, acc3, false, false);
    }

    // epilogue: C layout -> m = tM*16 + 8*hi + v ; n = tN*64 + sub*16 + l15
    const int m0 = tM * 16 + hi * 8;
    v8f accs[4] = {acc0, acc1, acc2, acc3};
    #pragma unroll
    for (int sub = 0; sub < 4; ++sub) {
        const int n = tN * 64 + sub * 16 + l15;
        const float bsv = bias[n];
        #pragma unroll
        for (int v = 0; v < 8; ++v) {
            const size_t idx = (size_t)(m0 + v) * N + n;
            float val = accs[sub][v] + bsv;
            if (mode == 0)      outF[idx] += val;
            else if (mode == 1) outH[idx]  = f2b(val);
            else                outH[idx]  = f2b(gelu_tanh(val));
        }
    }
}

// ============================================================================
// Kernel 6: dozer sparse attention. One block per (bc, head).
// Q/K/V head-slices staged into LDS with GLOBAL_LOAD_ASYNC_TO_LDS_B128 (DMA,
// ASYNCcnt-fenced); scores + mask + softmax + A@V computed in-block.
// Single __shared__ block -> LDS byte offsets are known from base 0.
// Mask = local window | strided | hashed-random (approximation of host RNG).
// ============================================================================
__global__ __launch_bounds__(256)
void dozer_attn(const unsigned short* __restrict__ Qh,
                const unsigned short* __restrict__ Kh,
                const unsigned short* __restrict__ Vh,
                unsigned short* __restrict__ Oh,
                int Lq, int Lk, int seed) {
    using namespace cfg;
    // layout (u16 units): qs[28*192] | ks[30*192] | vs[30*192] | ss (28*32 f32)
    constexpr int QS0 = 0;
    constexpr int KS0 = 28 * 192;
    constexpr int VS0 = KS0 + 30 * 192;
    constexpr int SS0 = VS0 + 30 * 192;            // f32 area, 4B aligned
    __shared__ __align__(16) unsigned short smem[SS0 + 28 * 32 * 2];
    unsigned short* qs = smem + QS0;
    unsigned short* ks = smem + KS0;
    unsigned short* vs = smem + VS0;
    float* ss = (float*)(smem + SS0);

    const int bc = blockIdx.x / Hh;
    const int h  = blockIdx.x % Hh;
    const int tid = threadIdx.x, lane = tid & 31, wid = tid >> 5;
    const int colBase = h * Dh;
    constexpr int CPR = Dh / 8;                    // 24 16-byte chunks per row

    // ---- async DMA stage: global -> LDS, 16B per op, no VGPR data path ----
    for (int t = tid; t < Lq * CPR; t += 256) {
        int i = t / CPR, ch = t % CPR;
        async_to_lds_b128((unsigned)((QS0 + i * Dh + ch * 8) * 2),
                          Qh + ((size_t)(bc * Lq + i)) * Dm + colBase + ch * 8);
    }
    for (int t = tid; t < Lk * CPR; t += 256) {
        int j = t / CPR, ch = t % CPR;
        async_to_lds_b128((unsigned)((KS0 + j * Dh + ch * 8) * 2),
                          Kh + ((size_t)(bc * Lk + j)) * Dm + colBase + ch * 8);
        async_to_lds_b128((unsigned)((VS0 + j * Dh + ch * 8) * 2),
                          Vh + ((size_t)(bc * Lk + j)) * Dm + colBase + ch * 8);
    }
    wait_asynccnt0();
    __syncthreads();

    // ---- scores + dozer mask ----
    const float scale = rsqrtf((float)Dh);
    const int offset = Lk - Lq;
    for (int p = tid; p < Lq * Lk; p += 256) {
        int i = p / Lk, j = p % Lk;
        float acc = 0.f;
        #pragma unroll 8
        for (int d = 0; d < Dh; ++d)
            acc += b2f(qs[i * Dh + d]) * b2f(ks[j * Dh + d]);
        int rel = i + offset - j;
        bool keep = (rel <= LOCAL && rel >= -LOCAL) || ((j % STRIDE) == 0);
        unsigned u = (unsigned)(i * 1664525u) ^ (unsigned)(j * 1013904223u) ^ (unsigned)(seed * 69069u);
        u ^= u >> 16; u *= 2246822519u; u ^= u >> 13; u *= 3266489917u; u ^= u >> 16;
        keep = keep || ((u & 0xFFFFFFu) < 1677722u);   // ~0.1 rate
        ss[i * 32 + j] = keep ? acc * scale : -1e9f;
    }
    __syncthreads();

    // ---- softmax over keys + A@V ; wave w owns rows w, w+8, ... ----
    for (int i = wid; i < Lq; i += 8) {
        float v = (lane < Lk) ? ss[i * 32 + lane] : -3.4e38f;
        float mx = v;
        #pragma unroll
        for (int o = 16; o; o >>= 1) mx = fmaxf(mx, __shfl_xor(mx, o, 32));
        float e = (lane < Lk) ? __expf(v - mx) : 0.f;
        float sm = e;
        #pragma unroll
        for (int o = 16; o; o >>= 1) sm += __shfl_xor(sm, o, 32);
        float prob = e / sm;

        float acc[6] = {0.f, 0.f, 0.f, 0.f, 0.f, 0.f};
        for (int j = 0; j < Lk; ++j) {
            float aj = __shfl(prob, j, 32);
            #pragma unroll
            for (int dd = 0; dd < 6; ++dd)
                acc[dd] += aj * b2f(vs[j * Dh + dd * 32 + lane]);
        }
        unsigned short* orow = Oh + ((size_t)(bc * Lq + i)) * Dm + colBase;
        #pragma unroll
        for (int dd = 0; dd < 6; ++dd)
            orow[dd * 32 + lane] = f2b(acc[dd]);
    }
}

// ============================================================================
// Kernel 7: final rearrange back + identity (recomputed value-embedding)
// ============================================================================
__global__ __launch_bounds__(256)
void finalize_out(const float* __restrict__ X, const float* __restrict__ x_dec,
                  const float* __restrict__ val_w, const float* __restrict__ val_b,
                  float* __restrict__ out) {
    using namespace cfg;
    size_t t = (size_t)blockIdx.x * blockDim.x + threadIdx.x;
    const size_t total = (size_t)Bsz * Ee * Sdec * Pp * Cc;
    if (t >= total) return;
    size_t r = t;
    int c = (int)(r % Cc); r /= Cc;
    int p = (int)(r % Pp); r /= Pp;
    int s = (int)(r % Sdec); r /= Sdec;
    int e = (int)(r % Ee); r /= Ee;
    int b = (int)r;
    float xv = X[(((size_t)(b * Cc + c)) * Sdec + s) * Dm + p * Ee + e];
    float idn = x_dec[((size_t)b * Ldec + (s * Pp + p)) * Cc + c] * val_w[e] + val_b[e];
    out[t] = xv + idn;
}

// ============================================================================
// host: kernel_launch
// ============================================================================
extern "C" void kernel_launch(void* const* d_in, const int* in_sizes, int n_in,
                              void* d_out, int out_size, void* d_ws, size_t ws_size,
                              hipStream_t stream) {
    using namespace cfg;
    (void)in_sizes; (void)n_in; (void)out_size; (void)ws_size;

    const float* x_dec = (const float*)d_in[0];
    const float* cross = (const float*)d_in[1];
    const float* val_w = (const float*)d_in[2];
    const float* val_b = (const float*)d_in[3];
    const float* pos   = (const float*)d_in[4];
    const float* pre_g = (const float*)d_in[5];
    const float* pre_b = (const float*)d_in[6];
    const float* Wq = (const float*)d_in[7];  const float* bq = (const float*)d_in[8];
    const float* Wk = (const float*)d_in[9];  const float* bk = (const float*)d_in[10];
    const float* Wv = (const float*)d_in[11]; const float* bv = (const float*)d_in[12];
    const float* Wo = (const float*)d_in[13]; const float* bo = (const float*)d_in[14];
    const float* n1g = (const float*)d_in[15]; const float* n1b = (const float*)d_in[16];
    const float* n2g = (const float*)d_in[17]; const float* n2b = (const float*)d_in[18];
    const float* n3g = (const float*)d_in[19]; const float* n3b = (const float*)d_in[20];
    const float* W1 = (const float*)d_in[21]; const float* b1 = (const float*)d_in[22];
    const float* W2 = (const float*)d_in[23]; const float* b2 = (const float*)d_in[24];
    const float* fg = (const float*)d_in[25]; const float* fb = (const float*)d_in[26];
    float* out = (float*)d_out;

    // -------- workspace layout --------
    char* ws = (char*)d_ws;
    size_t off = 0;
    auto alloc = [&](size_t bytes) { size_t o = off; off = (off + bytes + 255) & ~(size_t)255; return o; };
    const size_t szW   = (size_t)Dm * Dm * 2;      // one DxD bf16 weight
    const size_t szW1  = (size_t)Dm * Fm * 2;
    size_t oWq0 = alloc(szW), oWk0 = alloc(szW), oWv0 = alloc(szW), oWo0 = alloc(szW);
    size_t oWq1 = alloc(szW), oWk1 = alloc(szW), oWv1 = alloc(szW), oWo1 = alloc(szW);
    size_t oW1  = alloc(szW1), oW2 = alloc(szW1);
    size_t oX   = alloc((size_t)Mx * Dm * 4);      // fp32 residual stream
    size_t oXh  = alloc((size_t)Mx * Dm * 2);      // bf16 shadow of LN(x)
    size_t oXCh = alloc((size_t)Mc * Dm * 2);      // bf16 rearranged cross
    size_t oQh  = alloc((size_t)Mx * Dm * 2);
    size_t oOh  = alloc((size_t)Mx * Dm * 2);
    size_t oKh  = alloc((size_t)Mc * Dm * 2);
    size_t oVh  = alloc((size_t)Mc * Dm * 2);
    size_t oGh  = oKh;                             // FFN intermediate aliases dead K/V (Mx*Fm*2 <= 2*Mc*Dm*2)

    float*          X   = (float*)(ws + oX);
    unsigned short* Xh  = (unsigned short*)(ws + oXh);
    unsigned short* XCh = (unsigned short*)(ws + oXCh);
    unsigned short* Qh  = (unsigned short*)(ws + oQh);
    unsigned short* Kh  = (unsigned short*)(ws + oKh);
    unsigned short* Vh  = (unsigned short*)(ws + oVh);
    unsigned short* Oh  = (unsigned short*)(ws + oOh);
    unsigned short* Gh  = (unsigned short*)(ws + oGh);

    auto gridFor = [](size_t n) { return (unsigned)((n + 255) / 256); };

    // -------- 1) convert+transpose all weights to bf16 (N,K) --------
    auto cvt = [&](const float* w, size_t o, int K, int N) {
        convert_wt<<<gridFor((size_t)K * N), 256, 0, stream>>>(w, (unsigned short*)(ws + o), K, N);
    };
    cvt(Wq,            oWq0, Dm, Dm);  cvt(Wq + (size_t)Dm * Dm, oWq1, Dm, Dm);
    cvt(Wk,            oWk0, Dm, Dm);  cvt(Wk + (size_t)Dm * Dm, oWk1, Dm, Dm);
    cvt(Wv,            oWv0, Dm, Dm);  cvt(Wv + (size_t)Dm * Dm, oWv1, Dm, Dm);
    cvt(Wo,            oWo0, Dm, Dm);  cvt(Wo + (size_t)Dm * Dm, oWo1, Dm, Dm);
    cvt(W1, oW1, Dm, Fm);              cvt(W2, oW2, Fm, Dm);

    // -------- 2) embeddings / rearranges --------
    build_x0<<<gridFor((size_t)Mx * Dm), 256, 0, stream>>>(x_dec, val_w, val_b, pos, X);
    build_xc<<<gridFor((size_t)Mc * Dm), 256, 0, stream>>>(cross, XCh);

    auto ln = [&](const float* g, const float* b, unsigned short* h) {
        layernorm_rows<<<Mx, 256, 0, stream>>>(X, g, b, h);
    };
    auto gemm = [&](const unsigned short* A, int M, int K, size_t oBt, int N,
                    const float* bias, float* oF, unsigned short* oH, int mode) {
        int waves = (M / 16) * (N / 64);
        gemm_bf16_wmma<<<(waves + 7) / 8, 256, 0, stream>>>(
            A, (const unsigned short*)(ws + oBt), bias, oF, oH, M, N, K, mode);
    };

    // -------- 3) pre-LN --------
    ln(pre_g, pre_b, Xh);

    // -------- 4) self-attention --------
    gemm(Xh, Mx, Dm, oWq0, Dm, bq, nullptr, Qh, 1);
    gemm(Xh, Mx, Dm, oWk0, Dm, bk, nullptr, Kh, 1);
    gemm(Xh, Mx, Dm, oWv0, Dm, bv, nullptr, Vh, 1);
    dozer_attn<<<BC * Hh, 256, 0, stream>>>(Qh, Kh, Vh, Oh, Sdec, Sdec, 1);
    gemm(Oh, Mx, Dm, oWo0, Dm, bo, X, nullptr, 0);          // X += O-proj (residual)

    // -------- 5) LN + cross-attention --------
    ln(n1g, n1b, Xh);
    gemm(Xh,  Mx, Dm, oWq1, Dm, bq + Dm, nullptr, Qh, 1);
    gemm(XCh, Mc, Dm, oWk1, Dm, bk + Dm, nullptr, Kh, 1);
    gemm(XCh, Mc, Dm, oWv1, Dm, bv + Dm, nullptr, Vh, 1);
    dozer_attn<<<BC * Hh, 256, 0, stream>>>(Qh, Kh, Vh, Oh, Sdec, Senc, 2);
    gemm(Oh, Mx, Dm, oWo1, Dm, bo + Dm, X, nullptr, 0);     // X += cross O-proj

    // -------- 6) LN + FFN --------
    ln(n2g, n2b, Xh);
    gemm(Xh, Mx, Dm, oW1, Fm, b1, nullptr, Gh, 2);          // gelu(x@W1+b1) -> bf16
    gemm(Gh, Mx, Fm, oW2, Dm, b2, X, nullptr, 0);           // X += FFN out

    // -------- 7) final double LN + rearrange-back + identity --------
    ln(n3g, n3b, nullptr);
    ln(fg,  fb,  nullptr);
    finalize_out<<<gridFor((size_t)Bsz * Ee * Sdec * Pp * Cc), 256, 0, stream>>>(
        X, x_dec, val_w, val_b, out);
}
